// AttentionBlock_86870008529204
// MI455X (gfx1250) — matrix operand order repaired and checked by reading, not compiled, not run on hardware
//
#include <hip/hip_runtime.h>

typedef __attribute__((ext_vector_type(16))) __bf16 v16bf;
typedef __attribute__((ext_vector_type(8)))  __bf16 v8bf;
typedef __attribute__((ext_vector_type(4)))  __bf16 v4bf;
typedef __attribute__((ext_vector_type(8)))  float  v8f;

#define DDIM 256
#define LDSTRIDE 264                       // bf16 elems per W row in LDS (+16B pad: conflict-free b128 reads)
#define SMEM_BYTES (2u * 256u * (unsigned)LDSTRIDE * 2u)   // Whi + Wlo planes = 270336 B

__device__ __forceinline__ float rsum16(float v) {
  // sum across the 16 lanes of a half-wave (rows live in lane groups [0..15], [16..31])
  v += __shfl_xor(v, 1, 32);
  v += __shfl_xor(v, 2, 32);
  v += __shfl_xor(v, 4, 32);
  v += __shfl_xor(v, 8, 32);
  return v;
}

__global__ __launch_bounds__(256)
void sparsemax_linear_kernel(const float* __restrict__ x,
                             const float* __restrict__ Wg,
                             float* __restrict__ out) {
  extern __shared__ __bf16 smem[];
  __bf16* Whi = smem;
  __bf16* Wlo = smem + 256 * LDSTRIDE;

  const int tid = threadIdx.x;

  // ---- Stage 1: cooperative load W (fp32) -> split bf16 planes in LDS ----
  {
    const float4* W4 = reinterpret_cast<const float4*>(Wg);
    for (int i = 0; i < 64; ++i) {
      int idx4 = tid + (i << 8);           // 16384 float4s total
      float4 w = W4[idx4];
      int base = idx4 << 2;
      int e = base >> 8;                   // output-feature row of W
      int d = base & 255;                  // input-feature col
      v4bf h, l;
      float f;
      f = w.x; h[0] = (__bf16)f; l[0] = (__bf16)(f - (float)h[0]);
      f = w.y; h[1] = (__bf16)f; l[1] = (__bf16)(f - (float)h[1]);
      f = w.z; h[2] = (__bf16)f; l[2] = (__bf16)(f - (float)h[2]);
      f = w.w; h[3] = (__bf16)f; l[3] = (__bf16)(f - (float)h[3]);
      *reinterpret_cast<v4bf*>(Whi + e * LDSTRIDE + d) = h;
      *reinterpret_cast<v4bf*>(Wlo + e * LDSTRIDE + d) = l;
    }
  }
  __syncthreads();

  // ---- Stage 2: per-wave 16-row GEMM tile, logits kept in 128 VGPRs ----
  const int wave     = tid >> 5;
  const int lane     = tid & 31;
  const int laneM    = lane & 15;
  const int laneHalf = lane >> 4;
  const int rowBase  = blockIdx.x * 128 + wave * 16;

  const float* xrow = x + (size_t)(rowBase + laneM) * DDIM;

  v8f acc[16] = {};                        // 16 N-tiles of C (16x16 f32 each)

  #pragma unroll
  for (int kt = 0; kt < 8; ++kt) {         // K = 256 in steps of 32
    // A fragment (16x32 bf16): lane holds K = k0..k0+7 and k0+16..k0+23
    const int k0 = kt * 32 + laneHalf * 8;
    float4 c0 = *reinterpret_cast<const float4*>(xrow + k0);
    float4 c1 = *reinterpret_cast<const float4*>(xrow + k0 + 4);
    float4 c2 = *reinterpret_cast<const float4*>(xrow + k0 + 16);
    float4 c3 = *reinterpret_cast<const float4*>(xrow + k0 + 20);
    v16bf ah, al;
    {
      float fv[16] = {c0.x, c0.y, c0.z, c0.w, c1.x, c1.y, c1.z, c1.w,
                      c2.x, c2.y, c2.z, c2.w, c3.x, c3.y, c3.z, c3.w};
      #pragma unroll
      for (int j = 0; j < 16; ++j) {
        __bf16 hv = (__bf16)fv[j];
        ah[j] = hv;
        al[j] = (__bf16)(fv[j] - (float)hv);
      }
    }

    // B fragments: lane (N = laneM) reads K = koff..koff+7 and koff+16..koff+23 of W row e from LDS
    const int koff = kt * 32 + laneHalf * 8;
    #pragma unroll
    for (int et = 0; et < 16; ++et) {
      const int e = et * 16 + laneM;
      const __bf16* ph = Whi + e * LDSTRIDE + koff;
      const __bf16* pl = Wlo + e * LDSTRIDE + koff;
      union { v16bf v; v8bf h[2]; } bh, bl;
      bh.h[0] = *reinterpret_cast<const v8bf*>(ph);
      bh.h[1] = *reinterpret_cast<const v8bf*>(ph + 16);
      bl.h[0] = *reinterpret_cast<const v8bf*>(pl);
      bl.h[1] = *reinterpret_cast<const v8bf*>(pl + 16);
      v8f c = acc[et];
      c = __builtin_amdgcn_wmma_f32_16x16x32_bf16(false, ah, false, bh.v, (short)0, c, false, false);
      c = __builtin_amdgcn_wmma_f32_16x16x32_bf16(false, al, false, bh.v, (short)0, c, false, false);
      c = __builtin_amdgcn_wmma_f32_16x16x32_bf16(false, ah, false, bl.v, (short)0, c, false, false);
      acc[et] = c;
    }
  }

  // ---- Stage 3: sparsemax per row via Michelot fixed-point (exact simplex proj.) ----
  // C layout: lane holds row m = v + 8*laneHalf at col = et*16 + laneM, for v=0..7
  float tauv[8];
  #pragma unroll
  for (int v = 0; v < 8; ++v) {
    float s = 0.0f;
    #pragma unroll
    for (int et = 0; et < 16; ++et) s += acc[et][v];
    s = rsum16(s);
    float tau = (s - 1.0f) * (1.0f / 256.0f);   // start with full active set
    for (int t = 0; t < 16; ++t) {
      float S = 0.0f, K = 0.0f;
      #pragma unroll
      for (int et = 0; et < 16; ++et) {
        float z = acc[et][v];
        if (z > tau) { S += z; K += 1.0f; }
      }
      S = rsum16(S);
      K = rsum16(K);
      tau = (S - 1.0f) / fmaxf(K, 1.0f);        // k>=1 invariant (max(z) > tau always)
    }
    tauv[v] = tau;
  }

  // ---- Stage 4: fused ReLU-threshold write-out ----
  #pragma unroll
  for (int v = 0; v < 8; ++v) {
    const int m = v + laneHalf * 8;
    float* p = out + (size_t)(rowBase + m) * DDIM + laneM;
    #pragma unroll
    for (int et = 0; et < 16; ++et) {
      p[et * 16] = fmaxf(acc[et][v] - tauv[v], 0.0f);
    }
  }
}

extern "C" void kernel_launch(void* const* d_in, const int* in_sizes, int n_in,
                              void* d_out, int out_size, void* d_ws, size_t ws_size,
                              hipStream_t stream) {
  (void)in_sizes; (void)n_in; (void)out_size; (void)d_ws; (void)ws_size;
  const float* x = (const float*)d_in[0];
  const float* W = (const float*)d_in[1];
  float* out = (float*)d_out;

  // allow >64KB dynamic LDS (idempotent, host-side, capture-safe)
  hipFuncSetAttribute(reinterpret_cast<const void*>(&sparsemax_linear_kernel),
                      hipFuncAttributeMaxDynamicSharedMemorySize, (int)SMEM_BYTES);

  const int nRows   = 262144;
  const int nBlocks = nRows / 128;   // 8 waves * 16 rows per block
  sparsemax_linear_kernel<<<nBlocks, 256, SMEM_BYTES, stream>>>(x, W, out);
}
